// LSTM_34720515621546
// MI455X (gfx1250) — matrix operand order repaired and checked by reading, not compiled, not run on hardware
//
#include <hip/hip_runtime.h>

typedef __attribute__((ext_vector_type(16))) _Float16 v16h;
typedef __attribute__((ext_vector_type(8)))  _Float16 v8h;
typedef __attribute__((ext_vector_type(8)))  float    v8f;

#define B_    256
#define T_    512
#define DIN_  256
#define H_    1024
#define DOUT_ 256
#define KTOT_ 1280   // H_ + DIN_ folded contraction

__device__ __forceinline__ float sig_(float x)  { return 1.0f / (1.0f + __expf(-x)); }
__device__ __forceinline__ float tanh_(float x) { return 2.0f / (1.0f + __expf(-2.0f * x)) - 1.0f; }

__device__ __forceinline__ v16h make_a(const _Float16* p) {
    // A-fragment: 16 contiguous K per lane, elems 0..15 -> K = base..base+15
    // (matches B-fragment contiguous order)
    v8h lo = *(const v8h*)(p);
    v8h hi = *(const v8h*)(p + 8);
    v16h a;
#pragma unroll
    for (int i = 0; i < 8; ++i) { a[i] = lo[i]; a[i + 8] = hi[i]; }
    return a;
}

// ---------------------------------------------------------------------------
// One LSTM timestep, fused: gates = h@U + x_t@W (+b), activations, c/h update.
// Wave tile: 16 rows (M) x 16 hidden units (n), computed for ALL 4 gates at
// columns {n, n+H, n+2H, n+3H} so the epilogue owns the full cell update.
// Block: 256 threads = 8 waves = 2(M) x 4(n) wave tiles -> 32 x 64 per block.
// Grid: (256/32, 1024/64) = (8, 16).
// ---------------------------------------------------------------------------
__global__ __launch_bounds__(256)
void lstm_step_kernel(const _Float16* __restrict__ xf16,   // [B][T][DIN] f16
                      const _Float16* __restrict__ bigB,   // [4H][KTOT] f16 (row j = unit j, cols: U^T|W^T)
                      const float* __restrict__ b_i, const float* __restrict__ b_f,
                      const float* __restrict__ b_c, const float* __restrict__ b_o,
                      const _Float16* __restrict__ hprev,  // [B][H] f16
                      _Float16* __restrict__ hnext,        // [B][H] f16
                      float* __restrict__ cstate,          // [B][H] f32 (in-place)
                      int t)
{
    const int lane  = threadIdx.x & 31;
    const int wave  = threadIdx.x >> 5;
    const int ha    = lane >> 4;      // lane half (0/1)
    const int l16   = lane & 15;
    const int wm    = wave & 1;
    const int wn    = wave >> 1;
    const int m0    = blockIdx.x * 32 + wm * 16;
    const int n0    = blockIdx.y * 64 + wn * 16;

    const int mrow  = m0 + l16;       // A row this lane feeds
    const int ncol  = n0 + l16;       // hidden unit (B^T row within each gate)

    const _Float16* arow_h = hprev + (size_t)mrow * H_;
    const _Float16* arow_x = xf16  + ((size_t)mrow * T_ + t) * DIN_;
    const _Float16* br0 = bigB + (size_t)(0 * H_ + ncol) * KTOT_;
    const _Float16* br1 = bigB + (size_t)(1 * H_ + ncol) * KTOT_;
    const _Float16* br2 = bigB + (size_t)(2 * H_ + ncol) * KTOT_;
    const _Float16* br3 = bigB + (size_t)(3 * H_ + ncol) * KTOT_;

    v8f acc0 = {}, acc1 = {}, acc2 = {}, acc3 = {};

    // --- recurrent part: K = 0..1023 over hprev / U^T ---
#pragma unroll 4
    for (int k0 = 0; k0 < H_; k0 += 32) {
        v16h a = make_a(arow_h + k0 + ha * 16);
        const int kb = k0 + ha * 16;  // B-fragment: 16 contiguous K per lane
        v16h b0 = *(const v16h*)(br0 + kb);
        v16h b1 = *(const v16h*)(br1 + kb);
        v16h b2 = *(const v16h*)(br2 + kb);
        v16h b3 = *(const v16h*)(br3 + kb);
        acc0 = __builtin_amdgcn_wmma_f32_16x16x32_f16(false, a, false, b0, (short)0, acc0, false, false);
        acc1 = __builtin_amdgcn_wmma_f32_16x16x32_f16(false, a, false, b1, (short)0, acc1, false, false);
        acc2 = __builtin_amdgcn_wmma_f32_16x16x32_f16(false, a, false, b2, (short)0, acc2, false, false);
        acc3 = __builtin_amdgcn_wmma_f32_16x16x32_f16(false, a, false, b3, (short)0, acc3, false, false);
    }
    // --- input projection part: K = 1024..1279 over x_t / W^T ---
#pragma unroll
    for (int k0 = 0; k0 < DIN_; k0 += 32) {
        v16h a = make_a(arow_x + k0 + ha * 16);
        const int kb = H_ + k0 + ha * 16;
        v16h b0 = *(const v16h*)(br0 + kb);
        v16h b1 = *(const v16h*)(br1 + kb);
        v16h b2 = *(const v16h*)(br2 + kb);
        v16h b3 = *(const v16h*)(br3 + kb);
        acc0 = __builtin_amdgcn_wmma_f32_16x16x32_f16(false, a, false, b0, (short)0, acc0, false, false);
        acc1 = __builtin_amdgcn_wmma_f32_16x16x32_f16(false, a, false, b1, (short)0, acc1, false, false);
        acc2 = __builtin_amdgcn_wmma_f32_16x16x32_f16(false, a, false, b2, (short)0, acc2, false, false);
        acc3 = __builtin_amdgcn_wmma_f32_16x16x32_f16(false, a, false, b3, (short)0, acc3, false, false);
    }

    // --- epilogue: C/D layout -> element (VGPR r, lane): M = m0+r+8*ha, N = ncol
    const float bi = b_i[ncol], bf = b_f[ncol], bg = b_c[ncol], bo = b_o[ncol];
#pragma unroll
    for (int r = 0; r < 8; ++r) {
        const int m = m0 + r + 8 * ha;
        const size_t idx = (size_t)m * H_ + ncol;
        const float ig = sig_(acc0[r] + bi);
        const float fg = sig_(acc1[r] + bf);
        const float gg = tanh_(acc2[r] + bg);
        const float og = sig_(acc3[r] + bo);
        const float cn = fg * cstate[idx] + ig * gg;
        cstate[idx] = cn;
        hnext[idx]  = (_Float16)(og * tanh_(cn));
    }
}

// ---------------------------------------------------------------------------
// Final projection: out[256x256] = h_T[256x1024] @ W_out + b_out  (f16 WMMA)
// Block: 8 waves = 2(M) x 4(N) wave tiles of 16x64 -> 32 x 256. Grid: 8.
// ---------------------------------------------------------------------------
__global__ __launch_bounds__(256)
void out_proj_kernel(const _Float16* __restrict__ hT,   // [B][H] f16
                     const _Float16* __restrict__ woT,  // [DOUT][H] f16 (W_out^T)
                     const float* __restrict__ b_out,
                     float* __restrict__ out)           // [B][DOUT] f32
{
    const int lane = threadIdx.x & 31;
    const int wave = threadIdx.x >> 5;
    const int ha   = lane >> 4;
    const int l16  = lane & 15;
    const int wm   = wave & 1;
    const int wn   = wave >> 1;
    const int m0   = blockIdx.x * 32 + wm * 16;

    const _Float16* arow = hT + (size_t)(m0 + l16) * H_;
    const int nc0 = wn * 64 + 0  + l16;
    const int nc1 = wn * 64 + 16 + l16;
    const int nc2 = wn * 64 + 32 + l16;
    const int nc3 = wn * 64 + 48 + l16;
    const _Float16* br0 = woT + (size_t)nc0 * H_;
    const _Float16* br1 = woT + (size_t)nc1 * H_;
    const _Float16* br2 = woT + (size_t)nc2 * H_;
    const _Float16* br3 = woT + (size_t)nc3 * H_;

    v8f acc0 = {}, acc1 = {}, acc2 = {}, acc3 = {};
#pragma unroll 4
    for (int k0 = 0; k0 < H_; k0 += 32) {
        v16h a = make_a(arow + k0 + ha * 16);
        const int kb = k0 + ha * 16;
        v16h b0 = *(const v16h*)(br0 + kb);
        v16h b1 = *(const v16h*)(br1 + kb);
        v16h b2 = *(const v16h*)(br2 + kb);
        v16h b3 = *(const v16h*)(br3 + kb);
        acc0 = __builtin_amdgcn_wmma_f32_16x16x32_f16(false, a, false, b0, (short)0, acc0, false, false);
        acc1 = __builtin_amdgcn_wmma_f32_16x16x32_f16(false, a, false, b1, (short)0, acc1, false, false);
        acc2 = __builtin_amdgcn_wmma_f32_16x16x32_f16(false, a, false, b2, (short)0, acc2, false, false);
        acc3 = __builtin_amdgcn_wmma_f32_16x16x32_f16(false, a, false, b3, (short)0, acc3, false, false);
    }

#pragma unroll
    for (int r = 0; r < 8; ++r) {
        const int m = m0 + r + 8 * ha;
        out[(size_t)m * DOUT_ + nc0] = acc0[r] + b_out[nc0];
        out[(size_t)m * DOUT_ + nc1] = acc1[r] + b_out[nc1];
        out[(size_t)m * DOUT_ + nc2] = acc2[r] + b_out[nc2];
        out[(size_t)m * DOUT_ + nc3] = acc3[r] + b_out[nc3];
    }
}

// ------------------------- prep kernels ------------------------------------
__global__ void conv_x_kernel(const float* __restrict__ x, _Float16* __restrict__ xf, int n4) {
    int i = (blockIdx.x * blockDim.x + threadIdx.x);
    if (i < n4) {
        float4 v = *(const float4*)(x + 4 * (size_t)i);
        _Float16* o = xf + 4 * (size_t)i;
        o[0] = (_Float16)v.x; o[1] = (_Float16)v.y; o[2] = (_Float16)v.z; o[3] = (_Float16)v.w;
    }
}

// bigB[j][k]: j = gate*H + n.  k<1024 -> U_gate[k][n];  k>=1024 -> W_gate[k-1024][n]
__global__ void pack_B_kernel(const float* __restrict__ Ui, const float* __restrict__ Uf,
                              const float* __restrict__ Uc, const float* __restrict__ Uo,
                              const float* __restrict__ Wi, const float* __restrict__ Wf,
                              const float* __restrict__ Wc, const float* __restrict__ Wo,
                              _Float16* __restrict__ bigB)
{
    const int j = blockIdx.y;                       // 0..4095 (uniform per block)
    const int k = blockIdx.x * 256 + threadIdx.x;   // 0..1279
    if (k >= KTOT_) return;
    const int g = j >> 10, n = j & 1023;
    const float* U = (g == 0) ? Ui : (g == 1) ? Uf : (g == 2) ? Uc : Uo;
    const float* W = (g == 0) ? Wi : (g == 1) ? Wf : (g == 2) ? Wc : Wo;
    const float v = (k < H_) ? U[(size_t)k * H_ + n] : W[(size_t)(k - H_) * H_ + n];
    bigB[(size_t)j * KTOT_ + k] = (_Float16)v;
}

__global__ void pack_Wo_kernel(const float* __restrict__ Wout, _Float16* __restrict__ woT) {
    const int j = blockIdx.y;                       // 0..255 (output unit)
    const int k = blockIdx.x * 256 + threadIdx.x;   // 0..1023
    woT[(size_t)j * H_ + k] = (_Float16)Wout[(size_t)k * DOUT_ + j];
}

__global__ void init_state_kernel(float* __restrict__ c, _Float16* __restrict__ h0, int n) {
    int i = blockIdx.x * 256 + threadIdx.x;
    if (i < n) { c[i] = 0.0f; h0[i] = (_Float16)0.0f; }
}

// ---------------------------------------------------------------------------
extern "C" void kernel_launch(void* const* d_in, const int* in_sizes, int n_in,
                              void* d_out, int out_size, void* d_ws, size_t ws_size,
                              hipStream_t stream) {
    (void)in_sizes; (void)n_in; (void)out_size; (void)ws_size;
    const float* x    = (const float*)d_in[0];
    const float* W_i  = (const float*)d_in[1];
    const float* U_i  = (const float*)d_in[2];
    const float* b_i  = (const float*)d_in[3];
    const float* W_f  = (const float*)d_in[4];
    const float* U_f  = (const float*)d_in[5];
    const float* b_f  = (const float*)d_in[6];
    const float* W_c  = (const float*)d_in[7];
    const float* U_c  = (const float*)d_in[8];
    const float* b_c  = (const float*)d_in[9];
    const float* W_o  = (const float*)d_in[10];
    const float* U_o  = (const float*)d_in[11];
    const float* b_o  = (const float*)d_in[12];
    const float* Wout = (const float*)d_in[13];
    const float* bout = (const float*)d_in[14];
    float* out = (float*)d_out;

    // workspace carve-up (256B aligned)
    char* ws = (char*)d_ws;
    size_t off = 0;
    auto take = [&](size_t bytes) { char* p = ws + off; off = (off + bytes + 255) & ~(size_t)255; return p; };
    _Float16* xf16 = (_Float16*)take((size_t)B_ * T_ * DIN_ * 2);   // 64 MiB
    _Float16* bigB = (_Float16*)take((size_t)4 * H_ * KTOT_ * 2);   // 10 MiB
    _Float16* woT  = (_Float16*)take((size_t)DOUT_ * H_ * 2);       // 0.5 MiB
    _Float16* hb0  = (_Float16*)take((size_t)B_ * H_ * 2);          // ping
    _Float16* hb1  = (_Float16*)take((size_t)B_ * H_ * 2);          // pong
    float*    cst  = (float*)take((size_t)B_ * H_ * 4);

    // prep
    {
        const int n4 = (B_ * T_ * DIN_) / 4;
        conv_x_kernel<<<(n4 + 255) / 256, 256, 0, stream>>>(x, xf16, n4);
        pack_B_kernel<<<dim3((KTOT_ + 255) / 256, 4 * H_), 256, 0, stream>>>(
            U_i, U_f, U_c, U_o, W_i, W_f, W_c, W_o, bigB);
        pack_Wo_kernel<<<dim3(H_ / 256, DOUT_), 256, 0, stream>>>(Wout, woT);
        const int ns = B_ * H_;
        init_state_kernel<<<(ns + 255) / 256, 256, 0, stream>>>(cst, hb0, ns);
    }

    // recurrence: kernel boundary is the inter-timestep barrier; ping-pong h
    for (int t = 0; t < T_; ++t) {
        const _Float16* hp = (t & 1) ? hb1 : hb0;
        _Float16*       hn = (t & 1) ? hb0 : hb1;
        lstm_step_kernel<<<dim3(B_ / 32, H_ / 64), 256, 0, stream>>>(
            xf16, bigB, b_i, b_f, b_c, b_o, hp, hn, cst, t);
    }
    // T_ = 512 (even) -> final h lives in hb0
    out_proj_kernel<<<dim3(B_ / 32), 256, 0, stream>>>(hb0, woT, bout, out);
}
